// NCMultiAgentPolicy_62294205661513
// MI455X (gfx1250) — matrix operand order repaired and checked by reading, not compiled, not run on hardware
//
#include <hip/hip_runtime.h>
#include <hip/hip_bf16.h>
#include <math.h>

// Problem constants (from reference)
#define A_   128
#define T_   2048
#define NS_  64
#define NA_  16
#define NFC_ 64
#define NH_  64
#define NG_  256   // 4*NH
#define K3_  192   // 3*NS == 3*NFC

typedef __attribute__((ext_vector_type(16))) _Float16 v16h;
typedef __attribute__((ext_vector_type(8)))  _Float16 v8h;
typedef __attribute__((ext_vector_type(4)))  _Float16 v4h;
typedef __attribute__((ext_vector_type(8)))  float    v8f;

// CDNA5 WMMA 16-bit A-matrix (16x32): per lane, elements 0..7 = K base..base+7,
// elements 8..15 = K base+16..base+23, base = (lane/16)*8. M = lane%16.
__device__ __forceinline__ int frag_a_base(int lane) { return (lane >> 4) << 3; }
// CDNA5 WMMA 16-bit B-matrix (32x16): N=lane%16, same K order as A. Unused: contiguous K base (lane/16)*16.
__device__ __forceinline__ int frag_b_kbase(int lane) { return (lane >> 4) << 4; }

__device__ __forceinline__ v16h join8(v8h lo, v8h hi) {
    v16h r;
#pragma unroll
    for (int i = 0; i < 8; ++i) { r[i] = lo[i]; r[8 + i] = hi[i]; }
    return r;
}

__device__ __forceinline__ float sigf(float x) { return 1.0f / (1.0f + __expf(-x)); }

// ---------------------------------------------------------------------------
// Kernel 0: one-time f32 -> f16 conversion (weights & activations consumed by
// the WMMA kernels). Without this, every one of the 16384 k1 tiles would
// re-convert the same per-agent weights (128x redundant cvt + f32 bandwidth).
// ---------------------------------------------------------------------------
__global__ __launch_bounds__(256)
void k0_convert(const float* __restrict__ src, _Float16* __restrict__ dst, int n4)
{
    const int i = blockIdx.x * 256 + threadIdx.x;
    if (i < n4) {
        const float4 v = ((const float4*)src)[i];
        v4h o;
        o[0] = (_Float16)v.x; o[1] = (_Float16)v.y;
        o[2] = (_Float16)v.z; o[3] = (_Float16)v.w;
        ((v4h*)dst)[i] = o;
    }
}

// ---------------------------------------------------------------------------
// Kernel 1: hoist all T-parallel work out of the scan.
//   For each (agent a, 16-step tile): relu_x, relu_p via WMMA -> LDS (f16),
//   then G_pre[t, 0:256] = [relu_x|relu_p] @ Wih[:, 0:128]^T + bih + bhh.
// 128 threads = 4 waves; wave w owns feature cols [16w,16w+16) (phase A) and
// gate cols [64w, 64w+64) (phase B). All fragment loads are 16B/32B vector
// loads thanks to the two-contiguous-runs structure of the A layout.
// ---------------------------------------------------------------------------
__global__ __launch_bounds__(128)
void k1_precompute(const _Float16* __restrict__ obsh,  // [T][A][NS] f16
                   const _Float16* __restrict__ fpsh,  // [T][A][NA] f16
                   const int*      __restrict__ js,    // [A][2]
                   const _Float16* __restrict__ Wxh,   // [A][NFC][K3] f16
                   const float*    __restrict__ bx,    // [A][NFC]
                   const _Float16* __restrict__ Wph,   // [A][NFC][2*NA] f16
                   const float*    __restrict__ bp,    // [A][NFC]
                   const _Float16* __restrict__ Wihh,  // [A][NG][K3] f16
                   const float*    __restrict__ bih,   // [A][NG]
                   const float*    __restrict__ bhh,   // [A][NG]
                   _Float16*       __restrict__ gpre)  // [A][T][NG] f16
{
    __shared__ _Float16 s3[16][128];   // [t_local][relu_x(64) | relu_p(64)]

    const int blk  = blockIdx.x;
    const int a    = blk >> 7;        // / (T/16)
    const int t0   = (blk & 127) << 4;
    const int lane = threadIdx.x & 31;
    const int wv   = threadIdx.x >> 5;
    const int j0   = js[a * 2 + 0];
    const int j1   = js[a * 2 + 1];
    const int trow = t0 + (lane & 15);
    const int fcol = wv * 16 + (lane & 15);
    const int ab   = frag_a_base(lane);

    // -------- phase A: relu_x (K=192, 6 chunks) --------
    v8f accx = {};
    for (int kc = 0; kc < 6; ++kc) {
        const int K0 = kc * 32;
        const int b0 = K0 + ab;        // first run of 8 K-values
        const int b1 = b0 + 16;        // second run of 8 K-values
        const int s0 = (b0 < 64) ? a : ((b0 < 128) ? j0 : j1);
        const int s1 = (b1 < 64) ? a : ((b1 < 128) ? j0 : j1);
        const v8h lo = *(const v8h*)&obsh[((size_t)trow * A_ + s0) * NS_ + (b0 & 63)];
        const v8h hi = *(const v8h*)&obsh[((size_t)trow * A_ + s1) * NS_ + (b1 & 63)];
        const v16h af = join8(lo, hi);
        const v16h bf = join8(*(const v8h*)&Wxh[((size_t)a * NFC_ + fcol) * K3_ + K0 + ab], *(const v8h*)&Wxh[((size_t)a * NFC_ + fcol) * K3_ + K0 + ab + 16]);
        accx = __builtin_amdgcn_wmma_f32_16x16x32_f16(false, af, false, bf,
                                                      (short)0, accx, false, false);
    }
    {
        const float b = bx[a * NFC_ + fcol];
#pragma unroll
        for (int r = 0; r < 8; ++r) {
            float v = accx[r] + b;
            if (v < 0.f) v = 0.f;
            s3[r + ((lane >> 4) << 3)][fcol] = (_Float16)v;
        }
    }

    // -------- phase A: relu_p (K=32, 1 chunk) --------
    {
        // runs: [ab, ab+8) from fps[j0], [ab+16, ab+24) -> fps[j1] cols ab..
        const v8h lo = *(const v8h*)&fpsh[((size_t)trow * A_ + j0) * NA_ + ab];
        const v8h hi = *(const v8h*)&fpsh[((size_t)trow * A_ + j1) * NA_ + ab];
        const v16h af = join8(lo, hi);
        const v16h bf = join8(*(const v8h*)&Wph[((size_t)a * NFC_ + fcol) * (2 * NA_) + ab], *(const v8h*)&Wph[((size_t)a * NFC_ + fcol) * (2 * NA_) + ab + 16]);
        v8f accp = {};
        accp = __builtin_amdgcn_wmma_f32_16x16x32_f16(false, af, false, bf,
                                                      (short)0, accp, false, false);
        const float b = bp[a * NFC_ + fcol];
#pragma unroll
        for (int r = 0; r < 8; ++r) {
            float v = accp[r] + b;
            if (v < 0.f) v = 0.f;
            s3[r + ((lane >> 4) << 3)][NFC_ + fcol] = (_Float16)v;
        }
    }
    __syncthreads();

    // -------- phase B: G_pre tile = s3[16,128] @ Wih[:,0:128]^T --------
    v8f acc[4] = {{}, {}, {}, {}};
    for (int kc = 0; kc < 4; ++kc) {
        const int K0 = kc * 32;
        const int b0 = K0 + ab;
        const v8h lo = *(const v8h*)&s3[lane & 15][b0];
        const v8h hi = *(const v8h*)&s3[lane & 15][b0 + 16];
        const v16h af = join8(lo, hi);
#pragma unroll
        for (int n = 0; n < 4; ++n) {
            const int g = wv * 64 + n * 16 + (lane & 15);
            const v16h bf = join8(*(const v8h*)&Wihh[((size_t)a * NG_ + g) * K3_ + K0 + ab], *(const v8h*)&Wihh[((size_t)a * NG_ + g) * K3_ + K0 + ab + 16]);
            acc[n] = __builtin_amdgcn_wmma_f32_16x16x32_f16(false, af, false, bf,
                                                            (short)0, acc[n], false, false);
        }
    }
#pragma unroll
    for (int n = 0; n < 4; ++n) {
        const int g = wv * 64 + n * 16 + (lane & 15);
        const float b = bih[a * NG_ + g] + bhh[a * NG_ + g];
#pragma unroll
        for (int r = 0; r < 8; ++r) {
            const int t = t0 + r + ((lane >> 4) << 3);
            gpre[((size_t)a * T_ + t) * NG_ + g] = (_Float16)(acc[n][r] + b);
        }
    }
}

// ---------------------------------------------------------------------------
// Device-wide per-step barrier: atomic sense-reversing barrier (16 co-resident
// workgroups) + CDNA5 cluster barrier (s_barrier_signal/wait -3; NOP if not
// launched as a 16-WG cluster, real HW sync if it is).
// ---------------------------------------------------------------------------
__device__ __forceinline__ void grid_barrier(unsigned* bar, unsigned* gen_local)
{
    __threadfence();
    __syncthreads();
    if (threadIdx.x == 0) {
        const unsigned target = ++(*gen_local);
        const unsigned arrived = atomicAdd(&bar[0], 1u);
        if (arrived == gridDim.x - 1) {
            atomicExch(&bar[0], 0u);
            __threadfence();
            atomicAdd(&bar[1], 1u);
        } else {
            while (atomicAdd(&bar[1], 0u) < target) { __builtin_amdgcn_s_sleep(1); }
        }
    }
    __builtin_amdgcn_s_cluster_barrier();   // CDNA5: cluster-wide split barrier
    __syncthreads();
}

// ---------------------------------------------------------------------------
// Kernel 2: the serial scan. 16 workgroups x 8 waves, one wave per agent.
// Ring halo exchange of h via ping-pong global buffer; weights stream from L2
// (all weights = 45 MB, fully L2-resident in MI455X's 192 MB L2).
// ---------------------------------------------------------------------------
__global__ __launch_bounds__(256)
void k2_recurrent(const float* __restrict__ dones,   // [T]
                  const float* __restrict__ states,  // [A][2*NH]
                  const int*   __restrict__ js,      // [A][2]
                  const float* __restrict__ Wm,      // [A][NFC][2*NH]
                  const float* __restrict__ bm,      // [A][NFC]
                  const float* __restrict__ Wih,     // [A][NG][K3] (cols 128..191)
                  const float* __restrict__ Whh,     // [A][NG][NH]
                  const _Float16* __restrict__ gpre, // [A][T][NG]
                  float*    __restrict__ hbuf,       // [2][A][NH] ping-pong
                  _Float16* __restrict__ hs,         // [A][T][NH] f16
                  float*    __restrict__ out_states, // [A][2*NH]
                  unsigned* __restrict__ bar)        // [count, gen]
{
    __shared__ float lm [8][128];   // neighbor h concat (per agent)
    __shared__ float lhh[8][64];    // keep * h
    __shared__ float lrm[8][64];    // relu_m

    const int wv   = threadIdx.x >> 5;
    const int lane = threadIdx.x & 31;
    const int a    = blockIdx.x * 8 + wv;
    const int j0   = js[a * 2 + 0];
    const int j1   = js[a * 2 + 1];

    // init: h0 -> hbuf[0], c0 -> registers (lane owns indices {lane, lane+32})
    float c0 = states[a * (2 * NH_) + NH_ + lane];
    float c1 = states[a * (2 * NH_) + NH_ + lane + 32];
    hbuf[a * NH_ + lane]      = states[a * (2 * NH_) + lane];
    hbuf[a * NH_ + lane + 32] = states[a * (2 * NH_) + lane + 32];

    unsigned gen = 0;
    grid_barrier(bar, &gen);        // h0 visible everywhere before step 0

    int p = 0;
    for (int t = 0; t < T_; ++t) {
        const float keep = 1.0f - dones[t];
        const float* hb = hbuf + (size_t)p * A_ * NH_;

        // pull next step's gate precompute toward this WGP while we compute
        if (t + 1 < T_)
            __builtin_prefetch(&gpre[((size_t)a * T_ + t + 1) * NG_], 0, 1);

        // stage neighbor m (raw h) and own keep*h into LDS
        lm[wv][lane]       = hb[j0 * NH_ + lane];
        lm[wv][lane + 32]  = hb[j0 * NH_ + lane + 32];
        lm[wv][lane + 64]  = hb[j1 * NH_ + lane];
        lm[wv][lane + 96]  = hb[j1 * NH_ + lane + 32];
        lhh[wv][lane]      = hb[a * NH_ + lane] * keep;
        lhh[wv][lane + 32] = hb[a * NH_ + lane + 32] * keep;
        __syncthreads();

        // relu_m = relu(Wm @ m + bm): each lane does rows {lane, lane+32}
        float r0 = bm[a * NFC_ + lane];
        float r1 = bm[a * NFC_ + lane + 32];
        {
            const float4* w0 = (const float4*)&Wm[((size_t)a * NFC_ + lane) * (2 * NH_)];
            const float4* w1 = (const float4*)&Wm[((size_t)a * NFC_ + lane + 32) * (2 * NH_)];
            const float4* mm = (const float4*)lm[wv];
#pragma unroll 8
            for (int k = 0; k < 32; ++k) {
                const float4 mv = mm[k];
                const float4 a0 = w0[k], a1 = w1[k];
                r0 += a0.x * mv.x + a0.y * mv.y + a0.z * mv.z + a0.w * mv.w;
                r1 += a1.x * mv.x + a1.y * mv.y + a1.z * mv.z + a1.w * mv.w;
            }
        }
        lrm[wv][lane]      = fmaxf(r0, 0.f);
        lrm[wv][lane + 32] = fmaxf(r1, 0.f);
        __syncthreads();

        // gates: g = G_pre[t] + Wih[:,128:192]@relu_m + Whh@(keep*h)
        // lane owns rows q*64 + {lane, lane+32} for q=0..3 (i,f,g,o)
        float g[8];
#pragma unroll
        for (int q = 0; q < 4; ++q) {
#pragma unroll
            for (int jj = 0; jj < 2; ++jj) {
                const int row = q * NH_ + lane + jj * 32;
                float acc = (float)gpre[((size_t)a * T_ + t) * NG_ + row];
                const float4* wi = (const float4*)&Wih[((size_t)a * NG_ + row) * K3_ + 128];
                const float4* wh = (const float4*)&Whh[((size_t)a * NG_ + row) * NH_];
                const float4* rm = (const float4*)lrm[wv];
                const float4* hh = (const float4*)lhh[wv];
#pragma unroll 4
                for (int k = 0; k < 16; ++k) {
                    const float4 wiv = wi[k], rv = rm[k];
                    acc += wiv.x * rv.x + wiv.y * rv.y + wiv.z * rv.z + wiv.w * rv.w;
                    const float4 whv = wh[k], hv = hh[k];
                    acc += whv.x * hv.x + whv.y * hv.y + whv.z * hv.z + whv.w * hv.w;
                }
                g[q * 2 + jj] = acc;
            }
        }

        // LSTM pointwise + publish h into the other ping-pong buffer
        float* hbn = hbuf + (size_t)(1 - p) * A_ * NH_;
#pragma unroll
        for (int jj = 0; jj < 2; ++jj) {
            const float gi = g[0 + jj], gf = g[2 + jj], gg = g[4 + jj], go = g[6 + jj];
            const float cc = (jj ? c1 : c0) * keep;
            const float cn = sigf(gf) * cc + sigf(gi) * tanhf(gg);
            const float hn = sigf(go) * tanhf(cn);
            if (jj) c1 = cn; else c0 = cn;
            const int j = lane + jj * 32;
            hbn[a * NH_ + j] = hn;
            hs[((size_t)a * T_ + t) * NH_ + j] = (_Float16)hn;
        }

        grid_barrier(bar, &gen);
        p ^= 1;
    }

    // new_states = concat(hT, cT)
    const float* hbf = hbuf + (size_t)p * A_ * NH_;
    out_states[a * (2 * NH_) + lane]             = hbf[a * NH_ + lane];
    out_states[a * (2 * NH_) + lane + 32]        = hbf[a * NH_ + lane + 32];
    out_states[a * (2 * NH_) + NH_ + lane]       = c0;
    out_states[a * (2 * NH_) + NH_ + lane + 32]  = c1;
}

// ---------------------------------------------------------------------------
// Kernel 3: heads. logits = hs @ Wa^T via WMMA; log-softmax over NA=16 lives
// across 16 lanes of the accumulator -> shfl-xor reduction. Critic's one-hot
// GEMV collapses to two scalar lookups + a 64-length dot.
// ---------------------------------------------------------------------------
__global__ __launch_bounds__(128)
void k3_heads(const _Float16* __restrict__ hs,   // [A][T][NH]
              const int*   __restrict__ js,      // [A][2]
              const int*   __restrict__ acts,    // [A][T]
              const _Float16* __restrict__ Wah,  // [A][NA][NH] f16
              const float* __restrict__ ba,      // [A][NA]
              const float* __restrict__ Wc,      // [A][1][NH+2*NA]
              const float* __restrict__ bc,      // [A][1]
              float* __restrict__ out_logp,      // [A][T][NA]
              float* __restrict__ out_v)         // [A][T]
{
    const int wv   = threadIdx.x >> 5;
    const int lane = threadIdx.x & 31;
    const int tile = blockIdx.x * 4 + wv;      // 0 .. A*T/16-1
    const int a    = tile >> 7;
    const int t0   = (tile & 127) << 4;
    const int n    = lane & 15;
    const int ab   = frag_a_base(lane);
    const int trow = t0 + (lane & 15);

    v8f acc = {};
#pragma unroll
    for (int kc = 0; kc < 2; ++kc) {
        const int K0 = kc * 32;
        const int b0 = K0 + ab;
        const v8h lo = *(const v8h*)&hs[((size_t)a * T_ + trow) * NH_ + b0];
        const v8h hi = *(const v8h*)&hs[((size_t)a * T_ + trow) * NH_ + b0 + 16];
        const v16h af = join8(lo, hi);
        const v16h bf = join8(*(const v8h*)&Wah[((size_t)a * NA_ + n) * NH_ + K0 + ab], *(const v8h*)&Wah[((size_t)a * NA_ + n) * NH_ + K0 + ab + 16]);
        acc = __builtin_amdgcn_wmma_f32_16x16x32_f16(false, af, false, bf,
                                                     (short)0, acc, false, false);
    }
    const float bn = ba[a * NA_ + n];
#pragma unroll
    for (int r = 0; r < 8; ++r) {
        float x = acc[r] + bn;
        float m = x;
#pragma unroll
        for (int d = 1; d < 16; d <<= 1) m = fmaxf(m, __shfl_xor(m, d, 32));
        float s = __expf(x - m);
#pragma unroll
        for (int d = 1; d < 16; d <<= 1) s += __shfl_xor(s, d, 32);
        const float lp = x - (m + __logf(s));
        const int t = t0 + r + ((lane >> 4) << 3);
        out_logp[((size_t)a * T_ + t) * NA_ + n] = lp;
    }

    // critic: v = Wc[0:64].hs + Wc[64+act0] + Wc[80+act1] + bc
    {
        const int t  = t0 + (lane & 15);
        const int kh = (lane >> 4) * 32;
        const float* wc = &Wc[(size_t)a * (NH_ + 2 * NA_)];
        float part = 0.f;
#pragma unroll 8
        for (int k = 0; k < 32; ++k)
            part += wc[kh + k] * (float)hs[((size_t)a * T_ + t) * NH_ + kh + k];
        part += __shfl_xor(part, 16, 32);
        if (lane < 16) {
            const int jj0 = js[a * 2 + 0], jj1 = js[a * 2 + 1];
            const int a0 = acts[(size_t)jj0 * T_ + t];
            const int a1 = acts[(size_t)jj1 * T_ + t];
            out_v[(size_t)a * T_ + t] =
                part + wc[NH_ + a0] + wc[NH_ + NA_ + a1] + bc[a];
        }
    }
}

// ---------------------------------------------------------------------------
extern "C" void kernel_launch(void* const* d_in, const int* in_sizes, int n_in,
                              void* d_out, int out_size, void* d_ws, size_t ws_size,
                              hipStream_t stream)
{
    const float* obs    = (const float*)d_in[0];
    const float* fps    = (const float*)d_in[1];
    const float* dones  = (const float*)d_in[2];
    const int*   acts   = (const int*)  d_in[3];
    const float* states = (const float*)d_in[4];
    const int*   js     = (const int*)  d_in[5];
    const float* Wx  = (const float*)d_in[6];
    const float* bx  = (const float*)d_in[7];
    const float* Wp  = (const float*)d_in[8];
    const float* bp  = (const float*)d_in[9];
    const float* Wm  = (const float*)d_in[10];
    const float* bm  = (const float*)d_in[11];
    const float* Wih = (const float*)d_in[12];
    const float* bih = (const float*)d_in[13];
    const float* Whh = (const float*)d_in[14];
    const float* bhh = (const float*)d_in[15];
    const float* Wa  = (const float*)d_in[16];
    const float* ba  = (const float*)d_in[17];
    const float* Wc  = (const float*)d_in[18];
    const float* bc  = (const float*)d_in[19];

    const size_t nObs = (size_t)T_ * A_ * NS_;          // 16.8M
    const size_t nFps = (size_t)T_ * A_ * NA_;          // 4.2M
    const size_t nWx  = (size_t)A_ * NFC_ * K3_;        // 1.57M
    const size_t nWp  = (size_t)A_ * NFC_ * 2 * NA_;    // 0.26M
    const size_t nWih = (size_t)A_ * NG_ * K3_;         // 6.29M
    const size_t nWa  = (size_t)A_ * NA_ * NH_;         // 0.13M

    char* ws = (char*)d_ws;
    size_t off = 0;
    auto alloc_h = [&](size_t n) { _Float16* p = (_Float16*)(ws + off);
                                   off = (off + n * sizeof(_Float16) + 255) & ~(size_t)255; return p; };
    _Float16* gpre = alloc_h((size_t)A_ * T_ * NG_);
    _Float16* hsb  = alloc_h((size_t)A_ * T_ * NH_);
    _Float16* obsh = alloc_h(nObs);
    _Float16* fpsh = alloc_h(nFps);
    _Float16* Wxh  = alloc_h(nWx);
    _Float16* Wph  = alloc_h(nWp);
    _Float16* Wihh = alloc_h(nWih);
    _Float16* Wah  = alloc_h(nWa);
    float*    hbuf = (float*)(ws + off); off += ((size_t)2 * A_ * NH_ * sizeof(float) + 255) & ~(size_t)255;
    unsigned* bar  = (unsigned*)(ws + off); off += 256;

    float* out_logp   = (float*)d_out;
    float* out_v      = out_logp + (size_t)A_ * T_ * NA_;
    float* out_states = out_v + (size_t)A_ * T_;

    hipMemsetAsync(bar, 0, 256, stream);

    auto cvt = [&](const float* s, _Float16* d, size_t n) {
        const int n4 = (int)(n / 4);
        k0_convert<<<dim3((n4 + 255) / 256), dim3(256), 0, stream>>>(s, d, n4);
    };
    cvt(obs, obsh, nObs);
    cvt(fps, fpsh, nFps);
    cvt(Wx,  Wxh,  nWx);
    cvt(Wp,  Wph,  nWp);
    cvt(Wih, Wihh, nWih);
    cvt(Wa,  Wah,  nWa);

    k1_precompute<<<dim3(A_ * (T_ / 16)), dim3(128), 0, stream>>>(
        obsh, fpsh, js, Wxh, bx, Wph, bp, Wihh, bih, bhh, gpre);

    k2_recurrent<<<dim3(16), dim3(256), 0, stream>>>(
        dones, states, js, Wm, bm, Wih, Whh, gpre, hbuf, hsb, out_states, bar);

    k3_heads<<<dim3(A_ * (T_ / 16) / 4), dim3(128), 0, stream>>>(
        hsb, js, acts, Wah, ba, Wc, bc, out_logp, out_v);
}
